// EnhancedLNNManipulator_63350767616780
// MI455X (gfx1250) — matrix operand order repaired and checked by reading, not compiled, not run on hardware
//
#include <hip/hip_runtime.h>

typedef _Float16 f16;
typedef __attribute__((ext_vector_type(16))) _Float16 v16h;
typedef __attribute__((ext_vector_type(8)))  float    v8f;

#define DTT 0.01f

__constant__ float c_LOWER[6]  = {-6.28f,-6.28f,-3.14f,-6.28f,-6.28f,-6.28f};
__constant__ float c_UPPER[6]  = { 6.28f, 6.28f, 3.14f, 6.28f, 6.28f, 6.28f};
__constant__ float c_EFFORT[6] = {150.f,150.f,150.f,28.f,28.f,28.f};

// ---------------- LDS layout (bytes) ----------------
// weights (per block, f16, K padded to mult of 32, dead features zeroed)
constexpr int OFF_W1L = 0;                     // 128x32 f16
constexpr int OFF_W2L = OFF_W1L + 128*32*2;    // 128x128 f16
constexpr int OFF_W3L = OFF_W2L + 128*128*2;   // 32x128 f16 (rows>=21 zero)
constexpr int OFF_W1V = OFF_W3L + 32*128*2;    // 128x32 f16
constexpr int OFF_W2V = OFF_W1V + 128*32*2;    // 128x128 f16
constexpr int OFF_FLT = OFF_W2V + 128*128*2;   // float section
constexpr int OFF_B1L = OFF_FLT;               // 128 f32
constexpr int OFF_B2L = OFF_B1L + 128*4;       // 128 f32
constexpr int OFF_B3L = OFF_B2L + 128*4;       // 32  f32 (pad)
constexpr int OFF_B1V = OFF_B3L + 32*4;        // 128 f32
constexpr int OFF_B2V = OFF_B1V + 128*4;       // 128 f32
constexpr int OFF_W3V = OFF_B2V + 128*4;       // 128 f32
constexpr int SMEM_W_BYTES = OFF_W3V + 128*4;  // 92800

// per-wave scratch
constexpr int WOFF_TBUF  = 0;      // 16x32  f16
constexpr int WOFF_P0    = 1024;   // 16x128 f16
constexpr int WOFF_P1    = 5120;
constexpr int WOFF_P2    = 9216;
constexpr int WOFF_P3    = 13312;
constexpr int WOFF_TRIG  = 17408;  // 16x12  f32
constexpr int WOFF_YBUF  = 18176;  // 16x32  f32
constexpr int WOFF_DYBUF = 20224;  // 16x32  f32
constexpr int WOFF_S3    = 22272;  // 16x32  f16
constexpr int WOFF_GBUF  = 23296;  // 16x16  f32
constexpr int WAVE_BYTES = 24320;

constexpr int WAVES_PER_BLOCK = 4;
constexpr int ELEMS_PER_BLOCK = WAVES_PER_BLOCK * 16;   // 64
constexpr int SMEM_BYTES = SMEM_W_BYTES + WAVES_PER_BLOCK * WAVE_BYTES; // 190080

// ---------------- small math (hardware transcendentals) ----------------
// fused softplus + sigmoid from one v_exp_f32
__device__ __forceinline__ void act_pair(float x, float& h, float& s){
  const float e = __expf(x);                       // v_exp_f32
  const float r = __builtin_amdgcn_rcpf(1.f + e);  // v_rcp_f32
  const bool big = (x > 20.f);
  h = big ? x   : __logf(1.f + e);                 // v_log_f32
  s = big ? 1.f : e * r;
}
__device__ __forceinline__ float sg_fast(float x){
  return __builtin_amdgcn_rcpf(1.f + __expf(-x));
}
__device__ __forceinline__ v8f zero8(){ v8f z = {0.f,0.f,0.f,0.f,0.f,0.f,0.f,0.f}; return z; }

__device__ __forceinline__ v8f wmma_f16(v16h a, v16h b, v8f c){
  return __builtin_amdgcn_wmma_f32_16x16x32_f16(false, a, false, b, (short)0, c, false, false);
}

// A fragment: 16x32, row-major LDS array A[r*lda + k]. CDNA5 layout:
// lanes 0-15 row=lane K{0-7,16-23}; lanes 16-31 same rows K{8-15,24-31}.
__device__ __forceinline__ v16h load_a(const f16* A, int lda, int k0, int lane){
  const int r = lane & 15; const int koff = (lane >> 4) * 8;
  v16h a;
#pragma unroll
  for (int h = 0; h < 16; ++h){
    const int k = ((h < 8) ? h : (h + 8)) + koff;
    a[h] = A[r*lda + k0 + k];
  }
  return a;
}
// B fragment: 32x16. lanes 0-15 col=lane K{0-7,16-23} ; lanes 16-31 col=lane-16 K{8-15,24-31}.
// CM=true : B[k][n] = W[(n0+n)*ldw + k]   (weight stored out x in, forward pass)
// CM=false: B[k][n] = W[(k0+k)*ldw + n0+n] (plain row-major, backward pass)
template<bool CM>
__device__ __forceinline__ v16h load_b(const f16* W, int ldw, int n0, int k0, int lane){
  const int n = lane & 15; const int kh = (lane >> 4) * 8;
  v16h b;
#pragma unroll
  for (int h = 0; h < 16; ++h){
    const int k = k0 + kh + ((h < 8) ? h : (h + 8));
    b[h] = CM ? W[(n0 + n)*ldw + k] : W[k*ldw + n0 + n];
  }
  return b;
}

// forward layer: h = softplus(A @ W^T + b), s = sigmoid(pre); both f16 out (16 x NF)
template<int K, int NF>
__device__ __forceinline__ void fwd_layer(const f16* A, int lda, const f16* W, int ldw,
                                          const float* bias, f16* outh, f16* outs, int lane){
  const int hl = lane & 15, hi = lane >> 4;
#pragma unroll
  for (int nt = 0; nt < NF/16; ++nt){
    v8f c = zero8();
#pragma unroll
    for (int k0 = 0; k0 < K; k0 += 32)
      c = wmma_f16(load_a(A, lda, k0, lane), load_b<true>(W, ldw, nt*16, k0, lane), c);
    const int n = nt*16 + hl;
    const float bn = bias[n];
#pragma unroll
    for (int v = 0; v < 8; ++v){
      const int m = v + hi*8;
      float hx, sx;
      act_pair(c[v] + bn, hx, sx);
      outh[m*NF + n] = (f16)hx;
      outs[m*NF + n] = (f16)sx;
    }
  }
}

// mass layer 3: y (float, ld 32) and s3 (f16, ld 32)
__device__ __forceinline__ void fwd_layer3(const f16* A, const f16* W, const float* bias,
                                           float* ybuf, f16* s3buf, int lane){
  const int hl = lane & 15, hi = lane >> 4;
#pragma unroll
  for (int nt = 0; nt < 2; ++nt){
    v8f c = zero8();
#pragma unroll
    for (int k0 = 0; k0 < 128; k0 += 32)
      c = wmma_f16(load_a(A, 128, k0, lane), load_b<true>(W, 128, nt*16, k0, lane), c);
    const int n = nt*16 + hl;
    const float bn = bias[n];
#pragma unroll
    for (int v = 0; v < 8; ++v){
      const int m = v + hi*8;
      float hx, sx;
      act_pair(c[v] + bn, hx, sx);
      ybuf[m*32 + n]  = hx;
      s3buf[m*32 + n] = (f16)sx;
    }
  }
}

// tangent layer: out = sbuf * (A @ B), f16 out (16 x NF)
template<int K, int NF, bool CM>
__device__ __forceinline__ void jvp_layer(const f16* A, int lda, const f16* W, int ldw,
                                          const f16* sbuf, f16* outd, int lane){
  const int hl = lane & 15, hi = lane >> 4;
#pragma unroll
  for (int nt = 0; nt < NF/16; ++nt){
    v8f c = zero8();
#pragma unroll
    for (int k0 = 0; k0 < K; k0 += 32)
      c = wmma_f16(load_a(A, lda, k0, lane), load_b<CM>(W, ldw, nt*16, k0, lane), c);
    const int n = nt*16 + hl;
#pragma unroll
    for (int v = 0; v < 8; ++v){
      const int m = v + hi*8;
      outd[m*NF + n] = (f16)((float)sbuf[m*NF + n] * c[v]);
    }
  }
}

// tangent mass layer 3: dy (float, ld 32) = s3 * (A @ W3L^T)
__device__ __forceinline__ void jvp_layer3(const f16* A, const f16* W, const f16* s3buf,
                                           float* dybuf, int lane){
  const int hl = lane & 15, hi = lane >> 4;
#pragma unroll
  for (int nt = 0; nt < 2; ++nt){
    v8f c = zero8();
#pragma unroll
    for (int k0 = 0; k0 < 128; k0 += 32)
      c = wmma_f16(load_a(A, 128, k0, lane), load_b<true>(W, 128, nt*16, k0, lane), c);
    const int n = nt*16 + hl;
#pragma unroll
    for (int v = 0; v < 8; ++v){
      const int m = v + hi*8;
      dybuf[m*32 + n] = (float)s3buf[m*32 + n] * c[v];
    }
  }
}

// potential layer 2 fused with start of backward: d2 = sigmoid(pre2V) * W3V[n]
__device__ __forceinline__ void vlayer2(const f16* A, const f16* W, const float* bias,
                                        const float* w3v, f16* outd2, int lane){
  const int hl = lane & 15, hi = lane >> 4;
#pragma unroll
  for (int nt = 0; nt < 8; ++nt){
    v8f c = zero8();
#pragma unroll
    for (int k0 = 0; k0 < 128; k0 += 32)
      c = wmma_f16(load_a(A, 128, k0, lane), load_b<true>(W, 128, nt*16, k0, lane), c);
    const int n = nt*16 + hl;
    const float bn = bias[n], wn = w3v[n];
#pragma unroll
    for (int v = 0; v < 8; ++v){
      const int m = v + hi*8;
      outd2[m*128 + n] = (f16)(sg_fast(c[v] + bn) * wn);
    }
  }
}

// ---------------- weight staging ----------------
__device__ void stage_weights(char* smem,
    const float* W1L, const float* b1L, const float* W2L, const float* b2L,
    const float* W3L, const float* b3L, const float* W1V, const float* b1V,
    const float* W2V, const float* b2V, const float* W3V)
{
  f16* w1l = (f16*)(smem + OFF_W1L);
  f16* w2l = (f16*)(smem + OFF_W2L);
  f16* w3l = (f16*)(smem + OFF_W3L);
  f16* w1v = (f16*)(smem + OFF_W1V);
  f16* w2v = (f16*)(smem + OFF_W2V);
  float* fb1l = (float*)(smem + OFF_B1L);
  float* fb2l = (float*)(smem + OFF_B2L);
  float* fb3l = (float*)(smem + OFF_B3L);
  float* fb1v = (float*)(smem + OFF_B1V);
  float* fb2v = (float*)(smem + OFF_B2V);
  float* fw3v = (float*)(smem + OFF_W3V);
  const int nt = blockDim.x;
  for (int i = threadIdx.x; i < 128*32; i += nt){
    int o = i >> 5, k = i & 31;
    w1l[i] = (f16)((k < 12) ? W1L[o*12 + k] : 0.f);
    w1v[i] = (f16)((k < 12) ? W1V[o*12 + k] : 0.f);
  }
  for (int i = threadIdx.x; i < 128*128; i += nt){
    w2l[i] = (f16)W2L[i];
    w2v[i] = (f16)W2V[i];
  }
  for (int i = threadIdx.x; i < 32*128; i += nt){
    int o = i >> 7, k = i & 127;
    w3l[i] = (f16)((o < 21) ? W3L[o*128 + k] : 0.f);
  }
  for (int i = threadIdx.x; i < 128; i += nt){
    fb1l[i] = b1L[i]; fb2l[i] = b2L[i]; fb1v[i] = b1V[i]; fb2v[i] = b2V[i]; fw3v[i] = W3V[i];
  }
  for (int i = threadIdx.x; i < 32; i += nt) fb3l[i] = (i < 21) ? b3L[i] : 0.f;
}

// ---------------- per-stage accel kernel ----------------
__global__ __launch_bounds__(128) void accel_stage_kernel(
    const float* __restrict__ obs, const float* __restrict__ action,
    const float* W1L, const float* b1L, const float* W2L, const float* b2L,
    const float* W3L, const float* b3L, const float* W1V, const float* b1V,
    const float* W2V, const float* b2V, const float* W3V,
    float* __restrict__ kqd_ws, const int* __restrict__ violg,
    float* __restrict__ out, int Btot, int stage)
{
  extern __shared__ char smem[];
  stage_weights(smem, W1L,b1L,W2L,b2L,W3L,b3L,W1V,b1V,W2V,b2V,W3V);

  const int tid = threadIdx.x;
  const int lane = tid & 31, wv = tid >> 5;
  const int hl = lane & 15;

  const f16* W1Ls = (const f16*)(smem + OFF_W1L);
  const f16* W2Ls = (const f16*)(smem + OFF_W2L);
  const f16* W3Ls = (const f16*)(smem + OFF_W3L);
  const f16* W1Vs = (const f16*)(smem + OFF_W1V);
  const f16* W2Vs = (const f16*)(smem + OFF_W2V);
  const float* fb1L = (const float*)(smem + OFF_B1L);
  const float* fb2L = (const float*)(smem + OFF_B2L);
  const float* fb3L = (const float*)(smem + OFF_B3L);
  const float* fb1V = (const float*)(smem + OFF_B1V);
  const float* fb2V = (const float*)(smem + OFF_B2V);
  const float* fW3V = (const float*)(smem + OFF_W3V);

  char* wb = smem + SMEM_W_BYTES + wv * WAVE_BYTES;
  f16*   tbuf  = (f16*)(wb + WOFF_TBUF);
  f16*   P0    = (f16*)(wb + WOFF_P0);
  f16*   P1    = (f16*)(wb + WOFF_P1);
  f16*   P2    = (f16*)(wb + WOFF_P2);
  f16*   P3    = (f16*)(wb + WOFF_P3);
  float* trigf = (float*)(wb + WOFF_TRIG);
  float* ybuf  = (float*)(wb + WOFF_YBUF);
  float* dybuf = (float*)(wb + WOFF_DYBUF);
  f16*   s3buf = (f16*)(wb + WOFF_S3);
  float* gbuf  = (float*)(wb + WOFF_GBUF);

  const int e = blockIdx.x * ELEMS_PER_BLOCK + wv * 16 + hl;

  float q0[6], qd0[6], tauv[6], qe[6], qde[6];
  if (lane < 16){
#pragma unroll
    for (int j = 0; j < 6; ++j){
      q0[j]  = obs[e*12 + j];
      qd0[j] = obs[e*12 + 6 + j];
      tauv[j]= action[e*6 + j] * c_EFFORT[j];
    }
#pragma unroll
    for (int j = 0; j < 6; ++j){
      float k0v = (stage == 1 || stage == 2) ? kqd_ws[((size_t)0*Btot + e)*6 + j] : 0.f;
      float k1v = (stage >= 2) ? kqd_ws[((size_t)1*Btot + e)*6 + j] : 0.f;
      float k2v = (stage == 3) ? kqd_ws[((size_t)2*Btot + e)*6 + j] : 0.f;
      if (stage == 0)      { qe[j] = q0[j];                                      qde[j] = qd0[j]; }
      else if (stage == 1) { qe[j] = q0[j] + 0.5f*DTT*qd0[j];                    qde[j] = qd0[j] + 0.5f*DTT*k0v; }
      else if (stage == 2) { qe[j] = q0[j] + 0.5f*DTT*(qd0[j] + 0.5f*DTT*k0v);   qde[j] = qd0[j] + 0.5f*DTT*k1v; }
      else                 { qe[j] = q0[j] + DTT*(qd0[j] + 0.5f*DTT*k1v);        qde[j] = qd0[j] + DTT*k2v; }
    }
#pragma unroll
    for (int j = 0; j < 6; ++j){
      float cj = __cosf(qe[j]), sj = __sinf(qe[j]);   // v_cos_f32 / v_sin_f32
      trigf[hl*12 + 2*j]     = cj;
      trigf[hl*12 + 2*j + 1] = sj;
      tbuf[hl*32 + 2*j]      = (f16)cj;
      tbuf[hl*32 + 2*j + 1]  = (f16)sj;
    }
#pragma unroll
    for (int k = 12; k < 32; ++k) tbuf[hl*32 + k] = (f16)0.f;
  }
  __syncthreads();

  // ---- mass network forward ----
  fwd_layer<32,128>(tbuf, 32, W1Ls, 32, fb1L, P0, P1, lane);   // h1 -> P0, s1 -> P1
  __syncthreads();
  fwd_layer<128,128>(P0, 128, W2Ls, 128, fb2L, P2, P3, lane);  // h2 -> P2, s2 -> P3
  __syncthreads();
  fwd_layer3(P2, W3Ls, fb3L, ybuf, s3buf, lane);               // y -> ybuf, s3 -> s3buf
  __syncthreads();

  float Lm[21], w6[6], cacc[6], sd[6];
  if (lane < 16){
#pragma unroll
    for (int t = 0; t < 21; ++t) Lm[t] = ybuf[hl*32 + t];
#pragma unroll
    for (int j = 0; j < 6; ++j){
      float s = 0.f;
#pragma unroll
      for (int i = 0; i < 6; ++i) if (i >= j) s += Lm[i*(i+1)/2 + j] * qde[i];
      w6[j] = s; cacc[j] = 0.f; sd[j] = 0.f;
    }
  }
  __syncthreads();

  // ---- mass network JVP: 6 tangent directions ----
#pragma unroll
  for (int d = 0; d < 6; ++d){
    // layer-1 tangent is rank-1 sparse: dh1 = s1 * (-sin(q_d) W1L[:,2d] + cos(q_d) W1L[:,2d+1])
    for (int idx = lane; idx < 16*128; idx += 32){
      const int m = idx >> 7, n = idx & 127;
      const float cs = trigf[m*12 + 2*d];
      const float sn = trigf[m*12 + 2*d + 1];
      const float dpre = -sn * (float)W1Ls[n*32 + 2*d] + cs * (float)W1Ls[n*32 + 2*d + 1];
      P2[idx] = (f16)((float)P1[idx] * dpre);   // dh1 -> P2
    }
    __syncthreads();
    jvp_layer<128,128,true>(P2, 128, W2Ls, 128, P3, P0, lane); // dh2 = s2*(dh1@W2L^T) -> P0
    __syncthreads();
    jvp_layer3(P0, W3Ls, s3buf, dybuf, lane);                  // dy -> dybuf
    __syncthreads();
    if (lane < 16){
      float dLl[21];
#pragma unroll
      for (int t = 0; t < 21; ++t) dLl[t] = dybuf[hl*32 + t];
      float v1[6], u[6];
#pragma unroll
      for (int j = 0; j < 6; ++j){
        float s = 0.f;
#pragma unroll
        for (int i = 0; i < 6; ++i) if (i >= j) s += dLl[i*(i+1)/2 + j] * qde[i];
        v1[j] = s;
      }
#pragma unroll
      for (int i = 0; i < 6; ++i){
        float s = 0.f;
#pragma unroll
        for (int j = 0; j < 6; ++j) if (j <= i)
          s += dLl[i*(i+1)/2 + j] * w6[j] + Lm[i*(i+1)/2 + j] * v1[j];
        u[i] = s;
      }
      float dotu = 0.f;
#pragma unroll
      for (int i = 0; i < 6; ++i){ cacc[i] += qde[d]*u[i]; dotu += u[i]*qde[i]; }
      sd[d] = dotu;
    }
    __syncthreads();
  }

  // ---- potential network: forward + reverse gradient ----
  fwd_layer<32,128>(tbuf, 32, W1Vs, 32, fb1V, P2, P1, lane);   // h1V -> P2, s1V -> P1
  __syncthreads();
  vlayer2(P2, W2Vs, fb2V, fW3V, P0, lane);                     // d2 -> P0
  __syncthreads();
  jvp_layer<128,128,false>(P0, 128, W2Vs, 128, P1, P3, lane);  // d1 = s1V*(d2@W2V) -> P3
  __syncthreads();
  { // dt = d1 @ W1V  (N=16, cols 12..15 are zero-padded)
    v8f c = zero8();
#pragma unroll
    for (int k0 = 0; k0 < 128; k0 += 32)
      c = wmma_f16(load_a(P3, 128, k0, lane), load_b<false>(W1Vs, 32, 0, k0, lane), c);
    const int hi = lane >> 4;
#pragma unroll
    for (int v = 0; v < 8; ++v) gbuf[(v + hi*8)*16 + hl] = c[v];
  }
  __syncthreads();

  // ---- per-element epilogue: gravity chain, constraints, Coriolis, solve ----
  if (lane < 16){
    float g[6];
#pragma unroll
    for (int j = 0; j < 6; ++j)
      g[j] = gbuf[hl*16 + 2*j] * (-trigf[hl*12 + 2*j + 1])
           + gbuf[hl*16 + 2*j + 1] * trigf[hl*12 + 2*j];

    float rhs[6];
#pragma unroll
    for (int j = 0; j < 6; ++j){
      const float lo = c_LOWER[j] + 0.1f, up = c_UPPER[j] - 0.1f;
      float fj;
      if (violg[j])
        fj = (qe[j] <= lo) ? c_EFFORT[j] : ((qe[j] >= up) ? -c_EFFORT[j] : 0.f);
      else
        fj = -5.f * (1.f/(qe[j] - lo) - 1.f/(up - qe[j]));
      const float cj = cacc[j] - 0.5f*sd[j];
      rhs[j] = tauv[j] - cj - g[j] - fj;
    }
    // solve (L L^T) a = rhs
    float yv[6], av[6];
#pragma unroll
    for (int i = 0; i < 6; ++i){
      float s = rhs[i];
#pragma unroll
      for (int j = 0; j < 6; ++j) if (j < i) s -= Lm[i*(i+1)/2 + j] * yv[j];
      yv[i] = s / Lm[i*(i+1)/2 + i];
    }
#pragma unroll
    for (int ii = 5; ii >= 0; --ii){
      float s = yv[ii];
#pragma unroll
      for (int j = 0; j < 6; ++j) if (j > ii) s -= Lm[j*(j+1)/2 + ii] * av[j];
      av[ii] = s / Lm[ii*(ii+1)/2 + ii];
    }
#pragma unroll
    for (int j = 0; j < 6; ++j) kqd_ws[((size_t)stage*Btot + e)*6 + j] = av[j];

    if (stage == 3){
#pragma unroll
      for (int j = 0; j < 6; ++j){
        const float k0v = kqd_ws[((size_t)0*Btot + e)*6 + j];
        const float k1v = kqd_ws[((size_t)1*Btot + e)*6 + j];
        const float k2v = kqd_ws[((size_t)2*Btot + e)*6 + j];
        const float k3v = av[j];
        float qn  = q0[j] + DTT*qd0[j] + (DTT*DTT/6.f)*(k0v + k1v + k2v);
        qn = fminf(fmaxf(qn, c_LOWER[j]), c_UPPER[j]);
        const float qdn = qd0[j] + (DTT/6.f)*(k0v + 2.f*k1v + 2.f*k2v + k3v);
        out[e*12 + j]     = qn;
        out[e*12 + 6 + j] = qdn;
      }
    }
  }
}

// ---------------- batch-wide violation reduction (jnp.any over axis=0) ----------------
__global__ void zero_viol_kernel(int* v){ if (threadIdx.x < 24) v[threadIdx.x] = 0; }

__global__ void viol_kernel(const float* __restrict__ obs, const float* __restrict__ kqd,
                            int* __restrict__ viol, int Btot, int stage){
  const int e = blockIdx.x * blockDim.x + threadIdx.x;
  if (e >= Btot) return;
#pragma unroll
  for (int j = 0; j < 6; ++j){
    const float q0  = obs[e*12 + j];
    const float qd0 = obs[e*12 + 6 + j];
    float qv;
    if (stage == 0)      qv = q0;
    else if (stage == 1) qv = q0 + 0.5f*DTT*qd0;
    else if (stage == 2) qv = q0 + 0.5f*DTT*(qd0 + 0.5f*DTT*kqd[((size_t)0*Btot + e)*6 + j]);
    else                 qv = q0 + DTT*(qd0 + 0.5f*DTT*kqd[((size_t)1*Btot + e)*6 + j]);
    const float lo = c_LOWER[j] + 0.1f, up = c_UPPER[j] - 0.1f;
    if (qv <= lo || qv >= up) atomicOr(&viol[j], 1);
  }
}

// ---------------- host launch ----------------
extern "C" void kernel_launch(void* const* d_in, const int* in_sizes, int n_in,
                              void* d_out, int out_size, void* d_ws, size_t ws_size,
                              hipStream_t stream)
{
  const float* obs    = (const float*)d_in[0];
  const float* action = (const float*)d_in[1];
  const float* W1L = (const float*)d_in[2];  const float* b1L = (const float*)d_in[3];
  const float* W2L = (const float*)d_in[4];  const float* b2L = (const float*)d_in[5];
  const float* W3L = (const float*)d_in[6];  const float* b3L = (const float*)d_in[7];
  const float* W1V = (const float*)d_in[8];  const float* b1V = (const float*)d_in[9];
  const float* W2V = (const float*)d_in[10]; const float* b2V = (const float*)d_in[11];
  const float* W3V = (const float*)d_in[12];
  // d_in[13] = b3V: constant shift of the potential, gradient-free -> unused.

  const int Btot = in_sizes[0] / 12;
  float* kqd = (float*)d_ws;
  int* viol = (int*)((char*)d_ws + (size_t)4 * Btot * 6 * sizeof(float));

  zero_viol_kernel<<<1, 32, 0, stream>>>(viol);
  for (int s = 0; s < 4; ++s){
    viol_kernel<<<(Btot + 255)/256, 256, 0, stream>>>(obs, kqd, viol + s*6, Btot, s);
    accel_stage_kernel<<<Btot / ELEMS_PER_BLOCK, 128, SMEM_BYTES, stream>>>(
        obs, action, W1L,b1L,W2L,b2L,W3L,b3L,W1V,b1V,W2V,b2V,W3V,
        kqd, viol + s*6, (float*)d_out, Btot, s);
  }
  (void)n_in; (void)out_size; (void)ws_size;
}
